// FullRankRNN_11364483465790
// MI455X (gfx1250) — matrix operand order repaired and checked by reading, not compiled, not run on hardware
//
#include <hip/hip_runtime.h>

#define B_ 64
#define T_ 512
#define I_ 128
#define L_ 1024
#define O_ 128
#define NBLK 18u   // 16 h-update tiles + 2 output-projection tiles

typedef __attribute__((ext_vector_type(16))) __bf16        v16bf;
typedef __attribute__((ext_vector_type(8)))  float         v8f;
typedef __attribute__((ext_vector_type(4)))  unsigned int  v4u;
typedef __attribute__((ext_vector_type(8)))  unsigned int  v8u;

__device__ __forceinline__ unsigned short f2bf(float f) {
  unsigned int u = __float_as_uint(f);
  unsigned int r = u + 0x7FFFu + ((u >> 16) & 1u);   // round-to-nearest-even
  return (unsigned short)(r >> 16);
}

// A fragment (16-bit A 16x32): lane<16 -> K 0..15; lane>=16 -> K 16..31.
// p = row_base + kb + hh*16  (one contiguous 32-byte run)
__device__ __forceinline__ v16bf load_a_frag(const unsigned short* p) {
  v4u lo = *(const v4u*)p;
  v4u hi = *(const v4u*)(p + 8);
  v8u w = __builtin_shufflevector(lo, hi, 0, 1, 2, 3, 4, 5, 6, 7);
  return __builtin_bit_cast(v16bf, w);
}

// B fragment (16-bit B 32x16): lane<16 -> col=lane, K 0..15; lane>=16 -> col=lane-16, K 16..31.
// p = col_base + kb + hh*16  (one contiguous 32-byte run; ds_load_b128 when p is LDS)
__device__ __forceinline__ v16bf load_b_frag(const unsigned short* p) {
  v4u lo = *(const v4u*)p;
  v4u hi = *(const v4u*)(p + 8);
  v8u w = __builtin_shufflevector(lo, hi, 0, 1, 2, 3, 4, 5, 6, 7);
  return __builtin_bit_cast(v16bf, w);
}

__device__ __forceinline__ v8f wmma_bf16(v16bf a, v16bf b, v8f c) {
  return __builtin_amdgcn_wmma_f32_16x16x32_bf16(false, a, false, b, (short)0, c,
                                                 false, false);
}

// Device-wide split barrier: monotone counter, target = NBLK*(step+1). Only 18 blocks
// -> always co-resident, no deadlock risk. Release/acquire fences at agent scope make
// the double-buffered r/th global stores visible across workgroups.
__device__ __forceinline__ void grid_barrier(unsigned int* bar, unsigned int target) {
  __builtin_amdgcn_fence(__ATOMIC_RELEASE, "agent");
  __syncthreads();
  if (threadIdx.x == 0) {
    __hip_atomic_fetch_add(bar, 1u, __ATOMIC_RELAXED, __HIP_MEMORY_SCOPE_AGENT);
    while (__hip_atomic_load(bar, __ATOMIC_RELAXED, __HIP_MEMORY_SCOPE_AGENT) < target)
      __builtin_amdgcn_s_sleep(2);
  }
  __syncthreads();
  __builtin_amdgcn_fence(__ATOMIC_ACQUIRE, "agent");
}

// ---------------- prep kernels (bf16 conversion / transposes with gains folded in) ----

__global__ void k_cvt(const float* __restrict__ src, unsigned short* __restrict__ dst, int n) {
  int i = blockIdx.x * blockDim.x + threadIdx.x;
  if (i < n) dst[i] = f2bf(src[i]);
}

// wiT[l][i] = wi[i][l] * si[i]   (L_ x I_)
__global__ void k_wiT(const float* __restrict__ wi, const float* __restrict__ si,
                      unsigned short* __restrict__ wiT) {
  int i = blockIdx.x * blockDim.x + threadIdx.x;
  if (i >= I_ * L_) return;
  int l = i >> 7, k = i & (I_ - 1);
  wiT[i] = f2bf(wi[(size_t)k * L_ + l] * si[k]);
}

// woT[o][l] = wo[l][o] * so[o]   (O_ x L_)
__global__ void k_woT(const float* __restrict__ wo, const float* __restrict__ so,
                      unsigned short* __restrict__ woT) {
  int i = blockIdx.x * blockDim.x + threadIdx.x;
  if (i >= O_ * L_) return;
  int o = i >> 10, l = i & (L_ - 1);
  woT[i] = f2bf(wo[(size_t)l * O_ + o] * so[o]);
}

// h[b][l] = h0[l];  r0 = tanh(h0)  (matches reference r_init = tanh(h_init));  bar = 0
__global__ void k_init(const float* __restrict__ h0, float* __restrict__ h,
                       unsigned short* __restrict__ r0, unsigned int* __restrict__ bar) {
  int i = blockIdx.x * blockDim.x + threadIdx.x;
  if (i == 0) *bar = 0u;
  if (i >= B_ * L_) return;
  float v = h0[i & (L_ - 1)];
  h[i] = v;
  r0[i] = f2bf(tanhf(v));
}

// ---------------- persistent scan kernel ---------------------------------------------
// 18 blocks x 256 threads. Blocks 0..15: one 64x64 tile of the h-update (wrec slice in
// LDS). Blocks 16..17: 64x64 tile of the output projection (woT slice in LDS). One
// device barrier per time step replaces 512 kernel launches.

__global__ __launch_bounds__(256) void k_scan(
    const unsigned short* __restrict__ xbf,   // (B*T, I) bf16
    const unsigned short* __restrict__ wiT,   // (L, I)  bf16
    const unsigned short* __restrict__ wrecb, // (L, L)  bf16 (row n = B-column n)
    const unsigned short* __restrict__ woT,   // (O, L)  bf16
    const float* __restrict__ noise,          // (B,T,L)
    const float* __restrict__ bvec,           // (L)
    float* __restrict__ h,                    // (B,L) f32, in-place
    unsigned short* __restrict__ r0b, unsigned short* __restrict__ r1b,
    unsigned short* __restrict__ th0b, unsigned short* __restrict__ th1b,
    float* __restrict__ out, unsigned int* __restrict__ bar) {
  extern __shared__ unsigned short ldsw[];    // 64 cols x 1024 K bf16 = 128 KB

  const int blk = blockIdx.x;
  const int wave = threadIdx.x >> 5;
  const int lane = threadIdx.x & 31;
  const int hh = lane >> 4, lidx = lane & 15;
  const int mtile = wave & 3;
  const int nt0 = (wave >> 2) * 2;
  const bool is_out = (blk >= 16);
  const int nb_blk = is_out ? (blk - 16) * 64 : blk * 64;

  {  // stage this block's weight slice into LDS once (ds_store_b128)
    const unsigned short* src =
        is_out ? (woT + (size_t)nb_blk * L_) : (wrecb + (size_t)nb_blk * L_);
    for (int i = threadIdx.x * 8; i < 64 * L_; i += 256 * 8)
      *(v4u*)(ldsw + i) = *(const v4u*)(src + i);
  }
  __syncthreads();

  const int arow = mtile * 16 + lidx;     // A row (batch index 0..63)
  const int nl0 = nt0 * 16 + lidx;        // LDS-local B column
  const int nl1 = nl0 + 16;
  const int n0 = nb_blk + nl0;            // global column
  const int n1 = n0 + 16;
  const unsigned short* lb0 = ldsw + (size_t)nl0 * L_ + hh * 16;
  const unsigned short* lb1 = ldsw + (size_t)nl1 * L_ + hh * 16;

  if (!is_out) {
    const unsigned short* wb0 = wiT + (size_t)n0 * I_ + hh * 16;
    const unsigned short* wb1 = wiT + (size_t)n1 * I_ + hh * 16;
    const float bv0 = bvec[n0], bv1 = bvec[n1];
    for (int t = 0; t < T_; ++t) {
      const unsigned short* rin = (t & 1) ? r1b : r0b;
      unsigned short* rout = (t & 1) ? r0b : r1b;
      unsigned short* thout = (t & 1) ? th0b : th1b;
      v8f acc0 = {}, acc1 = {};
      {  // acc += r_prev @ wrec.T  (A from L2, B from LDS)
        const unsigned short* ap = rin + (size_t)arow * L_ + hh * 16;
#pragma unroll 4
        for (int kb = 0; kb < L_; kb += 32) {
          v16bf a = load_a_frag(ap + kb);
          acc0 = wmma_bf16(a, load_b_frag(lb0 + kb), acc0);
          acc1 = wmma_bf16(a, load_b_frag(lb1 + kb), acc1);
        }
      }
      {  // acc += x_t @ wi_full  (fuses the "u" precompute, K=128)
        const unsigned short* xp = xbf + ((size_t)arow * T_ + t) * I_ + hh * 16;
#pragma unroll
        for (int kb = 0; kb < I_; kb += 32) {
          v16bf a = load_a_frag(xp + kb);
          acc0 = wmma_bf16(a, load_b_frag(wb0 + kb), acc0);
          acc1 = wmma_bf16(a, load_b_frag(wb1 + kb), acc1);
        }
      }
      // h = 0.8h + 0.05*noise + 0.2*(r@wrec.T + u);  r' = tanh(h+b);  th' = tanh(h)
#pragma unroll
      for (int v = 0; v < 8; ++v) {
        int m = mtile * 16 + v + hh * 8;   // C/D: lane<16 -> M=v, lane>=16 -> M=v+8
        size_t i0 = (size_t)m * L_ + n0;
        size_t i1 = (size_t)m * L_ + n1;
        size_t nz = ((size_t)m * T_ + t) * L_;
        float h0n = 0.8f * h[i0] + 0.05f * noise[nz + n0] + 0.2f * acc0[v];
        float h1n = 0.8f * h[i1] + 0.05f * noise[nz + n1] + 0.2f * acc1[v];
        h[i0] = h0n;
        h[i1] = h1n;
        rout[i0] = f2bf(tanhf(h0n + bv0));
        rout[i1] = f2bf(tanhf(h1n + bv1));
        thout[i0] = f2bf(tanhf(h0n));
        thout[i1] = f2bf(tanhf(h1n));
      }
      grid_barrier(bar, NBLK * (unsigned)(t + 1));
    }
  } else {
    for (int t = 0; t < T_; ++t) {
      if (t > 0) {  // out_{t-1} = tanh(h_{t-1}) @ wo_full  (th written at step t-1)
        const unsigned short* thin = (t & 1) ? th1b : th0b;
        v8f acc0 = {}, acc1 = {};
        const unsigned short* ap = thin + (size_t)arow * L_ + hh * 16;
#pragma unroll 4
        for (int kb = 0; kb < L_; kb += 32) {
          v16bf a = load_a_frag(ap + kb);
          acc0 = wmma_bf16(a, load_b_frag(lb0 + kb), acc0);
          acc1 = wmma_bf16(a, load_b_frag(lb1 + kb), acc1);
        }
#pragma unroll
        for (int v = 0; v < 8; ++v) {
          int m = mtile * 16 + v + hh * 8;
          size_t row = ((size_t)m * T_ + (t - 1)) * O_;
          out[row + n0] = acc0[v];
          out[row + n1] = acc1[v];
        }
      }
      grid_barrier(bar, NBLK * (unsigned)(t + 1));
    }
    {  // final row: out_{511} from th[(511+1)&1] == th0b (visible after last barrier)
      v8f acc0 = {}, acc1 = {};
      const unsigned short* ap = th0b + (size_t)arow * L_ + hh * 16;
#pragma unroll 4
      for (int kb = 0; kb < L_; kb += 32) {
        v16bf a = load_a_frag(ap + kb);
        acc0 = wmma_bf16(a, load_b_frag(lb0 + kb), acc0);
        acc1 = wmma_bf16(a, load_b_frag(lb1 + kb), acc1);
      }
#pragma unroll
      for (int v = 0; v < 8; ++v) {
        int m = mtile * 16 + v + hh * 8;
        size_t row = ((size_t)m * T_ + (T_ - 1)) * O_;
        out[row + n0] = acc0[v];
        out[row + n1] = acc1[v];
      }
    }
  }
}

// ---------------- host ---------------------------------------------------------------

extern "C" void kernel_launch(void* const* d_in, const int* in_sizes, int n_in,
                              void* d_out, int out_size, void* d_ws, size_t ws_size,
                              hipStream_t stream) {
  (void)in_sizes; (void)n_in; (void)out_size; (void)ws_size;
  const float* input = (const float*)d_in[0];
  const float* noise = (const float*)d_in[1];
  const float* wi    = (const float*)d_in[2];
  const float* si    = (const float*)d_in[3];
  const float* wrec  = (const float*)d_in[4];
  const float* bvec  = (const float*)d_in[5];
  const float* wo    = (const float*)d_in[6];
  const float* so    = (const float*)d_in[7];
  const float* h0    = (const float*)d_in[8];
  float* out = (float*)d_out;

  char* ws = (char*)d_ws;
  unsigned short* xbf   = (unsigned short*)(ws + 0);         //  8 MB  (B*T, I) bf16
  unsigned short* wiT   = (unsigned short*)(ws + 8388608);   //  256KB (L, I)
  unsigned short* wrecb = (unsigned short*)(ws + 8650752);   //  2 MB  (L, L)
  unsigned short* woT   = (unsigned short*)(ws + 10747904);  //  256KB (O, L)
  float*          h     = (float*)(ws + 11010048);           //  256KB (B, L) f32
  unsigned short* r0b   = (unsigned short*)(ws + 11272192);  //  128KB
  unsigned short* r1b   = (unsigned short*)(ws + 11403264);  //  128KB
  unsigned short* th0b  = (unsigned short*)(ws + 11534336);  //  128KB
  unsigned short* th1b  = (unsigned short*)(ws + 11665408);  //  128KB
  unsigned int*   bar   = (unsigned int*)(ws + 11796480);    //  barrier counter

  k_cvt<<<(B_ * T_ * I_ + 255) / 256, 256, 0, stream>>>(input, xbf, B_ * T_ * I_);
  k_wiT<<<(I_ * L_ + 255) / 256, 256, 0, stream>>>(wi, si, wiT);
  k_cvt<<<(L_ * L_ + 255) / 256, 256, 0, stream>>>(wrec, wrecb, L_ * L_);
  k_woT<<<(O_ * L_ + 255) / 256, 256, 0, stream>>>(wo, so, woT);
  k_init<<<(B_ * L_ + 255) / 256, 256, 0, stream>>>(h0, h, r0b, bar);

  // 128 KB dynamic LDS per workgroup (WGP has 320 KB); raise the limit if needed.
  static const int kLds = 64 * L_ * 2;
  (void)hipFuncSetAttribute((const void*)k_scan,
                            hipFuncAttributeMaxDynamicSharedMemorySize, kLds);
  k_scan<<<NBLK, 256, kLds, stream>>>(xbf, wiT, wrecb, woT, noise, bvec, h,
                                      r0b, r1b, th0b, th1b, out, bar);
}
